// GlistaCell_68822555951197
// MI455X (gfx1250) — matrix operand order repaired and checked by reading, not compiled, not run on hardware
//
#include <hip/hip_runtime.h>
#include <hip/hip_bf16.h>

typedef __bf16 bf16_t;
typedef __attribute__((ext_vector_type(16))) __bf16 v16bf;
typedef __attribute__((ext_vector_type(8)))  __bf16 v8bf;
typedef __attribute__((ext_vector_type(8)))  float  v8f;
typedef __attribute__((ext_vector_type(4)))  unsigned int v4u;
typedef __attribute__((ext_vector_type(8)))  int v8i;
typedef __attribute__((ext_vector_type(4)))  int v4i;

constexpr int M_DIM = 512;
constexpr int N_DIM = 2048;
constexpr int B_DIM = 8192;
constexpr int CIN   = M_DIM + 2 * N_DIM;   // 4608 input cols
constexpr int COUT  = CIN + 2 * N_DIM;     // 8704 output cols
constexpr int XK_OFF = CIN - N_DIM;        // 2560
constexpr int CPREV_OFF = CIN - 2 * N_DIM; // 512
constexpr float THETA = 0.1f;
constexpr float STEP  = 0.9f;

// ---------------------------------------------------------------------------
// TDM: stage a (tile_rows x 32) bf16 tile (row stride = stride_elems) into LDS.
// Descriptor per CDNA5 ISA ch.8 (D# groups). data_size=1 (2 bytes), tile_dim0=32
// (contiguous K), tile_dim1=tile_rows, type=2 ("image"), count=1 user descriptor.
// ---------------------------------------------------------------------------
__device__ __forceinline__ void tdm_load_tile32(unsigned lds_off,
                                                const bf16_t* gsrc,
                                                unsigned tile_rows,
                                                unsigned tensor_dim0,
                                                unsigned tensor_dim1,
                                                unsigned stride_elems) {
  unsigned long long ga = (unsigned long long)(size_t)gsrc;
  v4u g0;
  g0[0] = 1u;                                        // count=1 (valid), user mode
  g0[1] = lds_off;                                   // lds_addr (bytes)
  g0[2] = (unsigned)(ga & 0xFFFFFFFFu);              // global_addr[31:0]
  g0[3] = (unsigned)((ga >> 32) & 0x1FFFFFFu) | (2u << 30); // addr[56:32] | type=2
  v8i g1;
  g1[0] = (int)(1u << 16);                           // data_size=1 -> 2 bytes
  g1[1] = (int)((tensor_dim0 & 0xFFFFu) << 16);      // tensor_dim0[15:0] @bits63:48
  g1[2] = (int)((tensor_dim0 >> 16) | ((tensor_dim1 & 0xFFFFu) << 16));
  g1[3] = (int)((tensor_dim1 >> 16) | (32u << 16));  // tile_dim0 = 32
  g1[4] = (int)tile_rows;                            // tile_dim1, tile_dim2 = 0
  g1[5] = (int)stride_elems;                         // tensor_dim0_stride[31:0]
  g1[6] = 0;                                         // stride hi / dim1_stride lo
  g1[7] = 0;
  v4i g2 = {0, 0, 0, 0};
  v4i g3 = {0, 0, 0, 0};
#if defined(__clang_major__) && (__clang_major__ >= 23)
  v8i g4 = {0, 0, 0, 0, 0, 0, 0, 0};
  __builtin_amdgcn_tensor_load_to_lds(g0, g1, g2, g3, g4, 0);
#else
  __builtin_amdgcn_tensor_load_to_lds(g0, g1, g2, g3, 0);
#endif
}

// ---------------------------------------------------------------------------
// Small prep kernels
// ---------------------------------------------------------------------------
__global__ void glista_copy_inputs(const float4* __restrict__ in,
                                   float4* __restrict__ out) {
  size_t i = (size_t)blockIdx.x * blockDim.x + threadIdx.x; // over B * 1152
  if (i < (size_t)B_DIM * (CIN / 4)) {
    size_t row = i / (CIN / 4), c = i % (CIN / 4);
    out[row * (COUT / 4) + c] = in[i];
  }
}

__global__ void glista_gx(const float* __restrict__ in, bf16_t* __restrict__ gxh) {
  size_t i = (size_t)blockIdx.x * blockDim.x + threadIdx.x; // over B*N
  if (i < (size_t)B_DIM * N_DIM) {
    size_t row = i / N_DIM, col = i % N_DIM;
    float x = in[row * CIN + XK_OFF + col];
    float g = (1.0f + 0.5f * __expf(-fabsf(x))) * x;
    gxh[i] = (bf16_t)g;
  }
}

__global__ void glista_conv_bf16(const float* __restrict__ src,
                                 bf16_t* __restrict__ dst, int n) {
  int i = blockIdx.x * blockDim.x + threadIdx.x;
  if (i < n) dst[i] = (bf16_t)src[i];
}

// Wt[n*512 + k] = W[k*2048 + n]  (so GEMM2's B-operand reads contiguously)
__global__ void glista_transpose_w(const float* __restrict__ W,
                                   bf16_t* __restrict__ Wt) {
  int i = blockIdx.x * blockDim.x + threadIdx.x; // over N*M
  if (i < N_DIM * M_DIM) {
    int n = i >> 9, k = i & 511;
    Wt[i] = (bf16_t)W[(size_t)k * N_DIM + n];
  }
}

// ---------------------------------------------------------------------------
// Shared GEMM core: 8 waves/block, each wave computes a 16x64 output strip
// (4 accumulator tiles) reusing one A fragment per K-step against 4 B tiles.
// Block's 64x32 bf16 B-panel is staged by the TDM, double-buffered so the DMA
// of panel k+32 overlaps the 4 WMMAs on panel k; one barrier per K-step.
// All 4 B fragments are fetched before the WMMA batch so the matrix pipe
// issues back-to-back behind a single dscnt drain.
//   Amat : row-major (B_DIM x K)
//   Bsrc : row-major (numCols x K)  -> Bmat[k, n] = Bsrc[n*K + k]
// ---------------------------------------------------------------------------
__device__ __forceinline__ void glista_gemm_core(const bf16_t* __restrict__ Amat,
                                                 const bf16_t* __restrict__ Bsrc,
                                                 bf16_t* sB, int K, int numCols,
                                                 int rowBase, int colBase,
                                                 int half, int l15, v8f acc[4]) {
  const bf16_t* aRow = Amat + (size_t)(rowBase + l15) * K;
  const unsigned ldsBase = (unsigned)(size_t)sB;
  const bf16_t* panelSrc = Bsrc + (size_t)colBase * K;

  if (threadIdx.x < 32) {
    tdm_load_tile32(ldsBase, panelSrc, 64, K, numCols, K);
    __builtin_amdgcn_s_wait_tensorcnt(0);
  }
  __syncthreads();

  int buf = 0;
  for (int k0 = 0; k0 < K; k0 += 32, buf ^= 1) {
    const bool hasNext = (k0 + 32) < K;
    if (threadIdx.x < 32 && hasNext) {
      tdm_load_tile32(ldsBase + (unsigned)(buf ^ 1) * (64 * 32 * 2),
                      panelSrc + (k0 + 32), 64, K, numCols, K);
    }
    // A fragment (contiguous k order, same as B): lane<16 -> K 0..15; lane>=16 -> K 16..31
    const int kA = k0 + half * 16;
    v8bf a0 = *(const v8bf*)(aRow + kA);
    v8bf a1 = *(const v8bf*)(aRow + kA + 8);
    v16bf a = __builtin_shufflevector(a0, a1, 0, 1, 2, 3, 4, 5, 6, 7,
                                      8, 9, 10, 11, 12, 13, 14, 15);
    __builtin_prefetch(aRow + kA + 32, 0, 3);
    const bf16_t* cur = sB + buf * (64 * 32);
    // Fetch all 4 B fragments first (one dscnt drain), then batch the WMMAs.
    v16bf b[4];
#pragma unroll
    for (int t = 0; t < 4; ++t) {
      // B fragment for cols [colBase+16t, +16): lane<16 -> col l15, K 0..15 ;
      // lane>=16 -> col l15, K 16..31 (panel row-major 64 x 32)
      b[t] = *(const v16bf*)(cur + (t * 16 + l15) * 32 + half * 16);
    }
#pragma unroll
    for (int t = 0; t < 4; ++t) {
      acc[t] = __builtin_amdgcn_wmma_f32_16x16x32_bf16(false, a, false, b[t],
                                                       (short)0, acc[t],
                                                       false, false);
    }
    if (threadIdx.x < 32 && hasNext) __builtin_amdgcn_s_wait_tensorcnt(0);
    __syncthreads();
  }
}

// GEMM1: res_bf16[b, m] = y[b, m] - sum_k gx[b,k] * A[m,k]     (K = 2048)
__global__ void __launch_bounds__(256)
glista_gemm1(const bf16_t* __restrict__ gxh, const bf16_t* __restrict__ Abf,
             const float* __restrict__ in, bf16_t* __restrict__ resh) {
  __shared__ __align__(32) bf16_t sB[2 * 64 * 32];
  const int wave = threadIdx.x >> 5;
  const int lane = threadIdx.x & 31;
  const int half = lane >> 4;
  const int l15  = lane & 15;
  const int colBase = blockIdx.x * 64;                 // over M=512 (8 groups)
  const int rowBase = (blockIdx.y * 8 + wave) * 16;    // over B=8192
  v8f acc[4] = {{0.f}, {0.f}, {0.f}, {0.f}};

  glista_gemm_core(gxh, Abf, sB, N_DIM, M_DIM, rowBase, colBase, half, l15, acc);

  // C layout: VGPR r -> M = r + half*8, N = l15
#pragma unroll
  for (int t = 0; t < 4; ++t) {
    const int col = colBase + t * 16 + l15;
#pragma unroll
    for (int r = 0; r < 8; ++r) {
      const int row = rowBase + half * 8 + r;
      float y = in[(size_t)row * CIN + col];
      resh[(size_t)row * M_DIM + col] = (bf16_t)(y - acc[t][r]);
    }
  }
}

// GEMM2: zk[b, n] = gx[b, n] + 0.9 * sum_k res[b,k] * W[k,n]   (K = 512)
__global__ void __launch_bounds__(256)
glista_gemm2(const bf16_t* __restrict__ resh, const bf16_t* __restrict__ Wt,
             const float* __restrict__ in, float* __restrict__ out) {
  __shared__ __align__(32) bf16_t sB[2 * 64 * 32];
  const int wave = threadIdx.x >> 5;
  const int lane = threadIdx.x & 31;
  const int half = lane >> 4;
  const int l15  = lane & 15;
  const int colBase = blockIdx.x * 64;                 // over N=2048 (32 groups)
  const int rowBase = (blockIdx.y * 8 + wave) * 16;    // over B=8192
  v8f acc[4] = {{0.f}, {0.f}, {0.f}, {0.f}};

  glista_gemm_core(resh, Wt, sB, M_DIM, N_DIM, rowBase, colBase, half, l15, acc);

#pragma unroll
  for (int t = 0; t < 4; ++t) {
    const int col = colBase + t * 16 + l15;
#pragma unroll
    for (int r = 0; r < 8; ++r) {
      const int row = rowBase + half * 8 + r;
      float xkv = in[(size_t)row * CIN + XK_OFF + col];
      float gx  = (1.0f + 0.5f * __expf(-fabsf(xkv))) * xkv;
      out[(size_t)row * COUT + CIN + N_DIM + col] = gx + STEP * acc[t][r];
    }
  }
}

// ---------------------------------------------------------------------------
// Finalize: per row, exact 103rd-largest |zk| (== percentile-95 'nearest' of
// 2048 elems) via 4-pass radix select on float bit patterns, then shrink+gate.
// ---------------------------------------------------------------------------
__global__ void __launch_bounds__(256)
glista_finalize(const float* __restrict__ in, float* __restrict__ out) {
  const int row = blockIdx.x;
  __shared__ float sAbs[N_DIM];
  __shared__ unsigned cnt[256];
  __shared__ unsigned sPrefix;
  __shared__ unsigned sRank;
  float* zrow = out + (size_t)row * COUT + CIN + N_DIM;

  for (int j = threadIdx.x; j < N_DIM; j += 256) sAbs[j] = fabsf(zrow[j]);
  if (threadIdx.x == 0) { sPrefix = 0u; sRank = 103u; } // 103rd largest
  __syncthreads();

  for (int shift = 24; shift >= 0; shift -= 8) {
    cnt[threadIdx.x] = 0u;
    __syncthreads();
    const unsigned pfx  = sPrefix;
    const unsigned mask = (shift == 24) ? 0u : (0xFFFFFFFFu << (shift + 8));
    for (int j = threadIdx.x; j < N_DIM; j += 256) {
      unsigned key = __float_as_uint(sAbs[j]);
      if ((key & mask) == pfx) atomicAdd(&cnt[(key >> shift) & 0xFFu], 1u);
    }
    __syncthreads();
    if (threadIdx.x == 0) {
      unsigned r = sRank, c = 0;
      int b = 255;
      for (; b > 0; --b) {
        if (c + cnt[b] >= r) break;
        c += cnt[b];
      }
      sRank = r - c;
      sPrefix = pfx | ((unsigned)b << shift);
    }
    __syncthreads();
  }
  const float thres = __uint_as_float(sPrefix);

  for (int j = threadIdx.x; j < N_DIM; j += 256) {
    float z  = zrow[j];
    float az = fabsf(z);
    float id = (az > THETA && az > thres) ? 1.0f : 0.0f;
    float ci = 1.0f - id;
    float soft = copysignf(fmaxf(az * ci - THETA, 0.0f), z);
    float xt = id * z + soft;
    float xkv   = in[(size_t)row * CIN + XK_OFF + j];
    float cprev = in[(size_t)row * CIN + CPREV_OFF + j];
    float p2 = THETA * cprev;
    float p1 = 1.0f / (fabsf(xt - xkv) + 0.1f);
    float g  = 1.0f - 0.5f * p1 * p2;
    out[(size_t)row * COUT + CIN + j] = ci;            // cindex
    zrow[j] = g * xt + (1.0f - g) * xkv;               // output (overwrites zk)
  }
}

// ---------------------------------------------------------------------------
extern "C" void kernel_launch(void* const* d_in, const int* in_sizes, int n_in,
                              void* d_out, int out_size, void* d_ws, size_t ws_size,
                              hipStream_t stream) {
  (void)in_sizes; (void)n_in; (void)out_size; (void)ws_size;
  const float* inputs = (const float*)d_in[0];
  const float* A      = (const float*)d_in[1];
  const float* W      = (const float*)d_in[2];
  float* out = (float*)d_out;

  char* ws = (char*)d_ws;
  bf16_t* gxh  = (bf16_t*)ws;                                          // B*N bf16
  bf16_t* resh = (bf16_t*)(ws + (size_t)B_DIM * N_DIM * 2);            // B*M bf16
  bf16_t* Abf  = (bf16_t*)(ws + (size_t)B_DIM * N_DIM * 2
                              + (size_t)B_DIM * M_DIM * 2);            // M*N bf16
  bf16_t* Wt   = (bf16_t*)(ws + (size_t)B_DIM * N_DIM * 2
                              + (size_t)B_DIM * M_DIM * 2
                              + (size_t)M_DIM * N_DIM * 2);            // N*M bf16

  // 1) copy inputs into out[:, :4608]
  {
    size_t n4 = (size_t)B_DIM * (CIN / 4);
    glista_copy_inputs<<<dim3((unsigned)((n4 + 255) / 256)), 256, 0, stream>>>(
        (const float4*)inputs, (float4*)out);
  }
  // 2) gx in bf16
  {
    size_t n = (size_t)B_DIM * N_DIM;
    glista_gx<<<dim3((unsigned)((n + 255) / 256)), 256, 0, stream>>>(inputs, gxh);
  }
  // 3) A -> bf16 ; W -> bf16 transposed
  glista_conv_bf16<<<dim3((M_DIM * N_DIM + 255) / 256), 256, 0, stream>>>(
      A, Abf, M_DIM * N_DIM);
  glista_transpose_w<<<dim3((M_DIM * N_DIM + 255) / 256), 256, 0, stream>>>(W, Wt);

  // 4) GEMM1: res = y - gx @ A^T  (8 col-groups x 64 row-blocks)
  glista_gemm1<<<dim3(M_DIM / 64, B_DIM / (16 * 8)), 256, 0, stream>>>(
      gxh, Abf, inputs, resh);

  // 5) GEMM2: zk = gx + 0.9 * res @ W   (32 col-groups x 64 row-blocks)
  glista_gemm2<<<dim3(N_DIM / 64, B_DIM / (16 * 8)), 256, 0, stream>>>(
      resh, Wt, inputs, out);

  // 6) per-row percentile + shrink + gate
  glista_finalize<<<dim3(B_DIM), 256, 0, stream>>>(inputs, out);
}
